// MetaPathGNN_19250043421008
// MI455X (gfx1250) — matrix operand order repaired and checked by reading, not compiled, not run on hardware
//
#include <hip/hip_runtime.h>
#include <hip/hip_bf16.h>

#define HH 128
#define OO 64
#define NN 100000
#define EE 800000

typedef __attribute__((ext_vector_type(16))) __bf16 v16bf;
typedef __attribute__((ext_vector_type(8)))  float  v8f;

// ---- workspace layout (bytes) ----
// [0      , 32768 )  WlT   : bf16[128][128]  (transposed Wl1, [n][k])
// [32768  , 65536 )  WcT   : bf16[128][128]  ((1-g)W0 + gW1, transposed)
// [65536  , 81920 )  WoutT : bf16[64][128]   (transposed Wout)
// [81920  , 82432 )  bc    : f32[128]        (combined bias)
// [82432  , 482432)  deg   : f32[N]
// [482432 , ...   )  agg   : f32[N][128]
#define WS_WLT   0
#define WS_WCT   32768
#define WS_WOUTT 65536
#define WS_BC    81920
#define WS_DEG   82432
#define WS_AGG   482432

#define WEIGHT_BYTES 81920          // WlT+WcT+WoutT contiguous
#define LNG_OFF      81920          // lng_s f32[128] in LDS
#define LNB_OFF      82432          // lnb_s f32[128] in LDS
#define WAVE_BASE    82944          // per-wave regions start here
#define SMEM_PER_WAVE (8192 + 256)  // y tile f32[16][128] + mu/rs/upd stats
#define WAVES_PER_BLOCK 4
#define SMEM_BYTES (WAVE_BASE + WAVES_PER_BLOCK * SMEM_PER_WAVE)

// ---------------- weight prep: transpose + gate-combine to bf16 -------------
__global__ void prep_weights(const float* __restrict__ Wl, const float* __restrict__ bl,
                             const float* __restrict__ W0, const float* __restrict__ b0,
                             const float* __restrict__ W1, const float* __restrict__ b1,
                             const float* __restrict__ gate, const float* __restrict__ Wout,
                             __bf16* __restrict__ WlT, __bf16* __restrict__ WcT,
                             __bf16* __restrict__ WoutT, float* __restrict__ bc) {
    float g = 1.0f / (1.0f + __expf(-gate[0]));
    int stride = gridDim.x * blockDim.x;
    for (int idx = blockIdx.x * blockDim.x + threadIdx.x; idx < HH * HH; idx += stride) {
        int k = idx >> 7, n = idx & 127;
        WlT[n * HH + k] = (__bf16)Wl[k * HH + n];
        WcT[n * HH + k] = (__bf16)((1.0f - g) * W0[k * HH + n] + g * W1[k * HH + n]);
        if (idx < OO * HH) {
            int ko = idx >> 6, no = idx & 63;
            WoutT[no * HH + ko] = (__bf16)Wout[ko * OO + no];
        }
        if (idx < HH) bc[idx] = bl[idx] + (1.0f - g) * b0[idx] + g * b1[idx];
    }
}

// ---------------- edge scatter: one wave32 per edge -------------------------
__global__ void edge_scatter(const float* __restrict__ xB,
                             const float* __restrict__ tB, const float* __restrict__ tC,
                             const int* __restrict__ edge2, const float* __restrict__ lam_raw,
                             float* __restrict__ agg, float* __restrict__ deg) {
    int lane  = threadIdx.x & 31;
    int warp  = (blockIdx.x * blockDim.x + threadIdx.x) >> 5;
    int nwarp = (gridDim.x * blockDim.x) >> 5;
    float lam = log1pf(__expf(lam_raw[0])) + 1e-8f;   // softplus + eps

    for (int e = warp; e < EE; e += nwarp) {
        int s = edge2[e];
        int d = edge2[EE + e];
        // prefetch next edge's source row (global_prefetch_b8)
        if (e + nwarp < EE) {
            int s2 = edge2[e + nwarp];
            __builtin_prefetch(xB + (size_t)s2 * HH + lane * 4, 0, 3);
        }
        float delta = fmaxf(tC[d] - tB[s], 0.0f);
        float w = __expf(fmaxf(-lam * delta, -60.0f));

        const float4 v = ((const float4*)(xB + (size_t)s * HH))[lane];
        float* dst = agg + (size_t)d * HH + lane * 4;
        atomicAdd(dst + 0, v.x * w);
        atomicAdd(dst + 1, v.y * w);
        atomicAdd(dst + 2, v.z * w);
        atomicAdd(dst + 3, v.w * w);
        if (lane == 0) atomicAdd(deg + d, w);
    }
}

// ---------------- WMMA helpers ----------------------------------------------
__device__ __forceinline__ v8f wmma_bf16(v16bf a, v16bf b, v8f c) {
    return __builtin_amdgcn_wmma_f32_16x16x32_bf16(false, a, false, b, (short)0, c, false, false);
}

// A fragment (16x32 bf16) for row m = lane&15, K-half hb = lane>>4,
// built from an f32 row (global or LDS), scaled, converted to bf16.
__device__ __forceinline__ v16bf a_frag_f32(const float* __restrict__ row, int kb, int hb, float scale) {
    const float* p = row + kb * 32 + hb * 16;
    v16bf a;
#pragma unroll
    for (int i = 0; i < 8; ++i) a[i]     = (__bf16)(p[i]      * scale);
#pragma unroll
    for (int i = 0; i < 8; ++i) a[8 + i] = (__bf16)(p[8 + i] * scale);
    return a;
}

// B fragment (32x16 bf16) from transposed weights W[n][k] in LDS:
// col n = lane&15 (+block), K-half hb = lane>>4 -> 16 contiguous bf16.
__device__ __forceinline__ v16bf b_frag(const __bf16* W, int n, int kb, int hb) {
    return *(const v16bf*)(W + n * HH + kb * 32 + hb * 16);
}

// ---------------- node update + LN + fused output GEMM ----------------------
__global__ void __launch_bounds__(WAVES_PER_BLOCK * 32)
node_update(const float* __restrict__ xC, const float* __restrict__ agg,
            const float* __restrict__ deg, const float* __restrict__ bc,
            const float* __restrict__ lng, const float* __restrict__ lnb,
            const float* __restrict__ bout,
            const __bf16* __restrict__ Wbase,   // WlT|WcT|WoutT contiguous in ws
            float* __restrict__ out) {
    extern __shared__ char smem[];

    // stage all bf16 weights + LN params into LDS once per block
    {
        const uint4* src = (const uint4*)Wbase;
        uint4* dst = (uint4*)smem;
        for (int i = threadIdx.x; i < WEIGHT_BYTES / 16; i += blockDim.x) dst[i] = src[i];
        float* lg = (float*)(smem + LNG_OFF);
        float* lb = (float*)(smem + LNB_OFF);
        for (int i = threadIdx.x; i < HH; i += blockDim.x) { lg[i] = lng[i]; lb[i] = lnb[i]; }
    }
    __syncthreads();

    const __bf16* WlT   = (const __bf16*)smem;
    const __bf16* WcT   = WlT + HH * HH;
    const __bf16* WoutT = WlT + 2 * HH * HH;
    const float*  lng_s = (const float*)(smem + LNG_OFF);
    const float*  lnb_s = (const float*)(smem + LNB_OFF);

    int wave = threadIdx.x >> 5;
    int lane = threadIdx.x & 31;
    float* y_s   = (float*)(smem + WAVE_BASE + wave * SMEM_PER_WAVE);
    float* mu_s  = y_s + 16 * HH;
    float* rs_s  = mu_s + 16;
    float* upd_s = rs_s + 16;

    int hb = lane >> 4;      // K-half select
    int nl = lane & 15;      // row (A) / col (B,D) within tile

    // per-lane bases with compile-time immediate offsets inside the loops
    float* ybase = y_s + (8 * hb) * HH + nl;        // + v*HH + nb*16

    int gw = blockIdx.x * WAVES_PER_BLOCK + wave;
    int nw = gridDim.x * WAVES_PER_BLOCK;
    const int ntiles = NN / 16;

    for (int tile = gw; tile < ntiles; tile += nw) {
        int node0 = tile * 16;
        const float* aRow  = agg + (size_t)(node0 + nl) * HH;
        const float* xRow  = xC  + (size_t)(node0 + nl) * HH;
        const float* xtile = xC  + (size_t)node0 * HH;  // xtile[idx], idx = r*128+c
        float* obase = out + (size_t)node0 * OO + (8 * hb) * OO + nl;  // + v*OO + ob*16

        float degv = deg[node0 + nl];
        float inv  = 1.0f / fmaxf(degv, 1e-6f);

        // hoist A fragments: invariant across all 8 output column-blocks
        v16bf aA[4], aX[4];
#pragma unroll
        for (int kb = 0; kb < 4; ++kb) {
            aA[kb] = a_frag_f32(aRow, kb, hb, inv);
            aX[kb] = a_frag_f32(xRow, kb, hb, 1.0f);
        }

        // y = (agg/deg)@Wl + xC@Wc + bc ; relu ; stage to LDS
#pragma unroll
        for (int nb = 0; nb < 8; ++nb) {
            v8f acc = {};
#pragma unroll
            for (int kb = 0; kb < 4; ++kb) {
                acc = wmma_bf16(aA[kb], b_frag(WlT, nb * 16 + nl, kb, hb), acc);
                acc = wmma_bf16(aX[kb], b_frag(WcT, nb * 16 + nl, kb, hb), acc);
            }
            float bias = bc[nb * 16 + nl];
#pragma unroll
            for (int v = 0; v < 8; ++v)
                ybase[v * HH + nb * 16] = fmaxf(acc[v] + bias, 0.0f);
        }
        __builtin_amdgcn_wave_barrier();

        // per-row layernorm statistics + update mask (lanes 0..15, one row each)
        if (lane < 16) {
            float s = 0.0f;
            for (int j = 0; j < HH; ++j) s += y_s[lane * HH + j];
            float mu = s * (1.0f / HH);
            float vs = 0.0f;
            for (int j = 0; j < HH; ++j) { float dd = y_s[lane * HH + j] - mu; vs += dd * dd; }
            mu_s[lane]  = mu;
            rs_s[lane]  = rsqrtf(vs * (1.0f / HH) + 1e-5f);
            upd_s[lane] = (degv > 0.0f) ? 1.0f : 0.0f;   // cnt>0 <=> deg>0
        }
        __builtin_amdgcn_wave_barrier();

        // normalize; rows with no incoming edges keep xC
#pragma unroll 4
        for (int t = 0; t < 64; ++t) {
            int idx = t * 32 + lane;          // idx = r*128 + c
            int r = idx >> 7, c = idx & 127;
            float val;
            if (upd_s[r] > 0.0f)
                val = (y_s[idx] - mu_s[r]) * rs_s[r] * lng_s[c] + lnb_s[c];
            else
                val = xtile[idx];
            y_s[idx] = val;
        }
        __builtin_amdgcn_wave_barrier();

        // fused output projection: out = h @ Wout + bout
        const float* hRow = y_s + nl * HH;
        v16bf aH[4];
#pragma unroll
        for (int kb = 0; kb < 4; ++kb) aH[kb] = a_frag_f32(hRow, kb, hb, 1.0f);

#pragma unroll
        for (int ob = 0; ob < 4; ++ob) {
            v8f acc = {};
#pragma unroll
            for (int kb = 0; kb < 4; ++kb)
                acc = wmma_bf16(aH[kb], b_frag(WoutT, ob * 16 + nl, kb, hb), acc);
            float bo = bout[ob * 16 + nl];
#pragma unroll
            for (int v = 0; v < 8; ++v)
                obase[v * OO + ob * 16] = acc[v] + bo;
        }
        __builtin_amdgcn_wave_barrier();
    }
}

// ---------------- host launch ------------------------------------------------
extern "C" void kernel_launch(void* const* d_in, const int* in_sizes, int n_in,
                              void* d_out, int out_size, void* d_ws, size_t ws_size,
                              hipStream_t stream) {
    (void)in_sizes; (void)n_in; (void)out_size; (void)ws_size;

    const float* xB    = (const float*)d_in[1];
    const float* xC    = (const float*)d_in[2];
    const float* tB    = (const float*)d_in[4];
    const float* tC    = (const float*)d_in[5];
    const int*   edge2 = (const int*)  d_in[7];
    const float* Wl1   = (const float*)d_in[18];
    const float* bl1   = (const float*)d_in[19];
    const float* W01   = (const float*)d_in[20];
    const float* b01   = (const float*)d_in[21];
    const float* W11   = (const float*)d_in[22];
    const float* b11   = (const float*)d_in[23];
    const float* gate1 = (const float*)d_in[24];
    const float* lam1  = (const float*)d_in[25];
    const float* lng1  = (const float*)d_in[26];
    const float* lnb1  = (const float*)d_in[27];
    const float* Wout  = (const float*)d_in[28];
    const float* bout  = (const float*)d_in[29];

    char* ws = (char*)d_ws;
    __bf16* WlT   = (__bf16*)(ws + WS_WLT);
    __bf16* WcT   = (__bf16*)(ws + WS_WCT);
    __bf16* WoutT = (__bf16*)(ws + WS_WOUTT);
    float*  bc    = (float*) (ws + WS_BC);
    float*  deg   = (float*) (ws + WS_DEG);
    float*  agg   = (float*) (ws + WS_AGG);

    // zero deg + agg (contiguous)
    hipMemsetAsync(ws + WS_DEG, 0, (size_t)(NN + (size_t)NN * HH) * sizeof(float), stream);

    prep_weights<<<64, 256, 0, stream>>>(Wl1, bl1, W01, b01, W11, b11, gate1, Wout,
                                         WlT, WcT, WoutT, bc);

    edge_scatter<<<4096, 256, 0, stream>>>(xB, tB, tC, edge2, lam1, agg, deg);

    node_update<<<512, WAVES_PER_BLOCK * 32, SMEM_BYTES, stream>>>(
        xC, agg, deg, bc, lng1, lnb1, bout, WlT, (float*)d_out);
}
